// LSTMModel_77360950936049
// MI455X (gfx1250) — matrix operand order repaired and checked by reading, not compiled, not run on hardware
//
#include <hip/hip_runtime.h>
#include <cstdint>
#include <cstddef>

// ---------------------------------------------------------------------------
// LSTM (B=256, T=512, I=512, H=1024) + FC(1024->512), bf16 WMMA w/ f32 accum
// Step kernel: A panel staged to LDS via global_load_async_to_lds_b128,
// B fragments 2-stage software-pipelined (no register rotation copies).
// ---------------------------------------------------------------------------

typedef __attribute__((ext_vector_type(16))) __bf16 v16bf;
typedef __attribute__((ext_vector_type(8)))  __bf16 bf16x8;
typedef __attribute__((ext_vector_type(8)))  float  v8f;

#define B_   256
#define T_   512
#define I_   512
#define H_   1024
#define G4_  4096          // 4*H
#define KT_  1536          // I + H (fused GEMM K)
#define O_   512
#define AROW 1544          // KT_ + 8 elem pad -> 3088B row stride (bank spread)

// ---- helpers --------------------------------------------------------------

__device__ __forceinline__ __bf16 f2bf(float f) {
    unsigned u = __builtin_bit_cast(unsigned, f);
    unsigned r = (u + 0x7FFFu + ((u >> 16) & 1u)) >> 16;
    unsigned short s = (unsigned short)r;
    return __builtin_bit_cast(__bf16, s);
}

__device__ __forceinline__ float sig_f(float x) {
    return 1.0f / (1.0f + __expf(-x));
}
__device__ __forceinline__ float tanh_f(float x) {
    return 2.0f / (1.0f + __expf(-2.0f * x)) - 1.0f;
}

// A fragment (16x32 bf16) from a row-major panel (global or LDS pointer).
// Lane l: m = l&15, half = l>>4; holds K = k0+8*half+{0..7} and +16..+23.
__device__ __forceinline__ v16bf load_afrag(const __bf16* __restrict__ base,
                                            int stride, int m0, int k0, int lane) {
    const int m = lane & 15, half = lane >> 4;
    const __bf16* p = base + (size_t)(m0 + m) * (size_t)stride + (size_t)(k0 + half * 8);
    bf16x8 lo = *(const bf16x8*)(p);
    bf16x8 hi = *(const bf16x8*)(p + 16);
    return __builtin_shufflevector(lo, hi, 0,1,2,3,4,5,6,7,8,9,10,11,12,13,14,15);
}

// B fragment: 32x16 bf16 of W^T; lane l holds column n=(l&15) == row (row0+n)
// of W (row stride `stride`), K = k0+8*(l>>4)+{0..7} and +16..+23.
__device__ __forceinline__ v16bf load_bfrag(const __bf16* __restrict__ W,
                                            int stride, int row0, int k0, int lane) {
    const int n = lane & 15, half = lane >> 4;
    const __bf16* p = W + (size_t)(row0 + n) * (size_t)stride + (size_t)(k0 + half * 8);
    bf16x8 lo = *(const bf16x8*)(p);
    bf16x8 hi = *(const bf16x8*)(p + 16);
    return __builtin_shufflevector(lo, hi, 0,1,2,3,4,5,6,7,8,9,10,11,12,13,14,15);
}

#define WMMA_BF16(A, Bm, C) \
    __builtin_amdgcn_wmma_f32_16x16x32_bf16(false, (A), false, (Bm), (short)0, (C), false, false)

// ---- prep kernels ---------------------------------------------------------

__global__ void pack_wz_kernel(const float* __restrict__ Wih,
                               const float* __restrict__ Whh,
                               __bf16* __restrict__ Wz) {
    size_t idx = (size_t)blockIdx.x * blockDim.x + threadIdx.x;
    if (idx >= (size_t)G4_ * KT_) return;
    int row = (int)(idx / KT_);
    int k   = (int)(idx % KT_);
    float v = (k < I_) ? Wih[(size_t)row * I_ + k]
                       : Whh[(size_t)row * H_ + (k - I_)];
    Wz[idx] = f2bf(v);
}

__global__ void pack_wfc_kernel(const float* __restrict__ Wfc, __bf16* __restrict__ o) {
    size_t idx = (size_t)blockIdx.x * blockDim.x + threadIdx.x;
    if (idx >= (size_t)O_ * H_) return;
    o[idx] = f2bf(Wfc[idx]);
}

__global__ void bias_fuse_kernel(const float* __restrict__ bi,
                                 const float* __restrict__ bh,
                                 float* __restrict__ o) {
    int idx = blockIdx.x * blockDim.x + threadIdx.x;
    if (idx >= G4_) return;
    o[idx] = bi[idx] + bh[idx];
}

__global__ void xconv_kernel(const float* __restrict__ x, __bf16* __restrict__ xbf) {
    size_t i0 = ((size_t)blockIdx.x * blockDim.x + threadIdx.x) * 8;
    if (i0 >= (size_t)B_ * T_ * I_) return;
    #pragma unroll
    for (int j = 0; j < 8; ++j) xbf[i0 + j] = f2bf(x[i0 + j]);
}

__global__ void zero_hc_kernel(__bf16* __restrict__ hA, __bf16* __restrict__ hB,
                               float* __restrict__ c) {
    size_t idx = (size_t)blockIdx.x * blockDim.x + threadIdx.x;
    if (idx >= (size_t)B_ * H_) return;
    hA[idx] = f2bf(0.0f);
    hB[idx] = f2bf(0.0f);
    c[idx]  = 0.0f;
}

// ---- per-timestep fused GEMM + gates --------------------------------------
// Block (8 waves): all waves share batch block m0 = (blockIdx>>3)*32; wave w
// owns hidden block n0 = ((blockIdx&7)*8 + w)*16 and all 4 gates of its
// 32x16 output. A panel [x_t | h] (32 x 1536 bf16) is staged to LDS with
// async-to-LDS loads; B (Wz) fragments use 2 alternating register buffers.
__global__ __launch_bounds__(256) void lstm_step_kernel(
        const __bf16* __restrict__ xbf_t,   // x bf16 at timestep t (batch stride T_*I_)
        const __bf16* __restrict__ h_in,    // [256][1024] bf16
        const __bf16* __restrict__ Wz,      // [4096][1536] bf16
        const float*  __restrict__ bias,    // [4096] = b_ih + b_hh
        float*        __restrict__ c,       // [256][1024] f32 (in/out)
        __bf16*       __restrict__ h_out)   // [256][1024] bf16
{
    __shared__ __bf16 As[32 * AROW];        // ~96.5 KB of the 320 KB WGP LDS

    const int lane = threadIdx.x & 31;
    const int w    = threadIdx.x >> 5;
    const int m0   = (blockIdx.x >> 3) * 32;
    const int n0   = ((blockIdx.x & 7) * 8 + w) * 16;

    // --- stage A panel: 32 rows x 1536 cols = 6144 x 16B chunks, async ---
    #pragma unroll
    for (int i = 0; i < 24; ++i) {
        const int cid  = (int)threadIdx.x + 256 * i;
        const int row  = cid / 192;            // 192 chunks per row
        const int col8 = (cid % 192) * 8;      // element column
        const __bf16* gp = (col8 < I_)
            ? (xbf_t + (size_t)(m0 + row) * (size_t)(T_ * I_) + col8)
            : (h_in  + (size_t)(m0 + row) * (size_t)H_ + (col8 - I_));
        const unsigned lds = (unsigned)(uintptr_t)(&As[row * AROW + col8]);
        asm volatile("global_load_async_to_lds_b128 %0, %1, off"
                     :: "v"(lds), "v"(gp) : "memory");
    }
    asm volatile("s_wait_asynccnt 0x0" ::: "memory");
    __syncthreads();

    v8f acc[4][2];
    #pragma unroll
    for (int g = 0; g < 4; ++g) {
        acc[g][0] = (v8f){0.f,0.f,0.f,0.f,0.f,0.f,0.f,0.f};
        acc[g][1] = (v8f){0.f,0.f,0.f,0.f,0.f,0.f,0.f,0.f};
    }

    // --- K loop over fused K=1536: 2-stage pipeline, no buffer copies ---
    v16bf b0[4], b1[4];
    #pragma unroll
    for (int g = 0; g < 4; ++g) b0[g] = load_bfrag(Wz, KT_, g * H_ + n0, 0, lane);
    #pragma unroll
    for (int g = 0; g < 4; ++g) b1[g] = load_bfrag(Wz, KT_, g * H_ + n0, 32, lane);

    for (int k = 0; k < KT_ - 64; k += 64) {
        {   // even chunk: consume b0, refill b0 for k+64
            const v16bf a0 = load_afrag(As, AROW, 0,  k, lane);
            const v16bf a1 = load_afrag(As, AROW, 16, k, lane);
            #pragma unroll
            for (int g = 0; g < 4; ++g) {
                acc[g][0] = WMMA_BF16(a0, b0[g], acc[g][0]);
                acc[g][1] = WMMA_BF16(a1, b0[g], acc[g][1]);
            }
            #pragma unroll
            for (int g = 0; g < 4; ++g)
                b0[g] = load_bfrag(Wz, KT_, g * H_ + n0, k + 64, lane);
        }
        {   // odd chunk: consume b1, refill b1 for k+96
            const v16bf a0 = load_afrag(As, AROW, 0,  k + 32, lane);
            const v16bf a1 = load_afrag(As, AROW, 16, k + 32, lane);
            #pragma unroll
            for (int g = 0; g < 4; ++g) {
                acc[g][0] = WMMA_BF16(a0, b1[g], acc[g][0]);
                acc[g][1] = WMMA_BF16(a1, b1[g], acc[g][1]);
            }
            #pragma unroll
            for (int g = 0; g < 4; ++g)
                b1[g] = load_bfrag(Wz, KT_, g * H_ + n0, k + 96, lane);
        }
    }
    {   // tail: chunk KT_-64 (b0), then KT_-32 (b1)
        const v16bf a0 = load_afrag(As, AROW, 0,  KT_ - 64, lane);
        const v16bf a1 = load_afrag(As, AROW, 16, KT_ - 64, lane);
        #pragma unroll
        for (int g = 0; g < 4; ++g) {
            acc[g][0] = WMMA_BF16(a0, b0[g], acc[g][0]);
            acc[g][1] = WMMA_BF16(a1, b0[g], acc[g][1]);
        }
    }
    {
        const v16bf a0 = load_afrag(As, AROW, 0,  KT_ - 32, lane);
        const v16bf a1 = load_afrag(As, AROW, 16, KT_ - 32, lane);
        #pragma unroll
        for (int g = 0; g < 4; ++g) {
            acc[g][0] = WMMA_BF16(a0, b1[g], acc[g][0]);
            acc[g][1] = WMMA_BF16(a1, b1[g], acc[g][1]);
        }
    }

    // --- epilogue: gates + cell update, fully in-register ---
    // C/D layout: vgpr r, lanes 0-15 -> M=r, lanes 16-31 -> M=r+8; N = lane&15.
    const int n   = lane & 15;
    const int mh  = lane >> 4;
    const int col = n0 + n;
    const float b_i = bias[0 * H_ + col];
    const float b_f = bias[1 * H_ + col];
    const float b_g = bias[2 * H_ + col];
    const float b_o = bias[3 * H_ + col];

    #pragma unroll
    for (int ms = 0; ms < 2; ++ms) {
        #pragma unroll
        for (int r = 0; r < 8; ++r) {
            const int m = m0 + ms * 16 + mh * 8 + r;
            const float ig = sig_f(acc[0][ms][r] + b_i);
            const float fg = sig_f(acc[1][ms][r] + b_f);
            const float gg = tanh_f(acc[2][ms][r] + b_g);
            const float og = sig_f(acc[3][ms][r] + b_o);
            const size_t idx = (size_t)m * H_ + col;
            const float cv = fg * c[idx] + ig * gg;
            c[idx] = cv;
            h_out[idx] = f2bf(og * tanh_f(cv));
        }
    }
}

// ---- final FC: out = h @ W_fc^T + b_fc ------------------------------------
__global__ __launch_bounds__(256) void fc_kernel(
        const __bf16* __restrict__ h,      // [256][1024] bf16
        const __bf16* __restrict__ Wfc,    // [512][1024] bf16
        const float*  __restrict__ bfc,    // [512]
        float*        __restrict__ out)    // [256][512]
{
    const int lane = threadIdx.x & 31;
    const int wv   = (int)((blockIdx.x * blockDim.x + threadIdx.x) >> 5);
    const int m0   = (wv >> 5) * 16;   // 0..15 -> batch rows
    const int j0   = (wv & 31) * 16;   // 0..31 -> output cols

    v8f acc = (v8f){0.f,0.f,0.f,0.f,0.f,0.f,0.f,0.f};
    v16bf a0 = load_afrag(h,   H_, m0, 0,  lane);
    v16bf b0 = load_bfrag(Wfc, H_, j0, 0,  lane);
    v16bf a1 = load_afrag(h,   H_, m0, 32, lane);
    v16bf b1 = load_bfrag(Wfc, H_, j0, 32, lane);
    for (int k = 0; k < H_ - 64; k += 64) {
        acc = WMMA_BF16(a0, b0, acc);
        a0 = load_afrag(h,   H_, m0, k + 64, lane);
        b0 = load_bfrag(Wfc, H_, j0, k + 64, lane);
        acc = WMMA_BF16(a1, b1, acc);
        a1 = load_afrag(h,   H_, m0, k + 96, lane);
        b1 = load_bfrag(Wfc, H_, j0, k + 96, lane);
    }
    acc = WMMA_BF16(a0, b0, acc);
    acc = WMMA_BF16(a1, b1, acc);

    const int j  = j0 + (lane & 15);
    const int mh = lane >> 4;
    const float bj = bfc[j];
    #pragma unroll
    for (int r = 0; r < 8; ++r) {
        out[(size_t)(m0 + mh * 8 + r) * O_ + j] = acc[r] + bj;
    }
}

// ---- host launch ----------------------------------------------------------

extern "C" void kernel_launch(void* const* d_in, const int* in_sizes, int n_in,
                              void* d_out, int out_size, void* d_ws, size_t ws_size,
                              hipStream_t stream) {
    const float* x    = (const float*)d_in[0];
    const float* Wih  = (const float*)d_in[1];
    const float* Whh  = (const float*)d_in[2];
    const float* bih  = (const float*)d_in[3];
    const float* bhh  = (const float*)d_in[4];
    const float* Wfc  = (const float*)d_in[5];
    const float* bfc  = (const float*)d_in[6];
    float* out = (float*)d_out;

    char* ws = (char*)d_ws;
    size_t off = 0;
    auto alloc = [&](size_t bytes) -> void* {
        void* p = ws + off;
        off += (bytes + 255) & ~(size_t)255;
        return p;
    };
    __bf16* xbf   = (__bf16*)alloc((size_t)B_ * T_ * I_ * 2);   // 128 MB
    __bf16* Wz    = (__bf16*)alloc((size_t)G4_ * KT_ * 2);      // 12 MB
    __bf16* Wfcb  = (__bf16*)alloc((size_t)O_ * H_ * 2);        // 1 MB
    float*  bias  = (float*) alloc((size_t)G4_ * 4);
    __bf16* hA    = (__bf16*)alloc((size_t)B_ * H_ * 2);
    __bf16* hB    = (__bf16*)alloc((size_t)B_ * H_ * 2);
    float*  cbuf  = (float*) alloc((size_t)B_ * H_ * 4);
    (void)ws_size; (void)in_sizes; (void)n_in; (void)out_size;

    // prep
    {
        size_t nwz = (size_t)G4_ * KT_;
        pack_wz_kernel<<<(unsigned)((nwz + 255) / 256), 256, 0, stream>>>(Wih, Whh, Wz);
        size_t nfc = (size_t)O_ * H_;
        pack_wfc_kernel<<<(unsigned)((nfc + 255) / 256), 256, 0, stream>>>(Wfc, Wfcb);
        bias_fuse_kernel<<<(G4_ + 255) / 256, 256, 0, stream>>>(bih, bhh, bias);
        size_t nx = (size_t)B_ * T_ * I_;
        xconv_kernel<<<(unsigned)((nx / 8 + 255) / 256), 256, 0, stream>>>(x, xbf);
        size_t nh = (size_t)B_ * H_;
        zero_hc_kernel<<<(unsigned)((nh + 255) / 256), 256, 0, stream>>>(hA, hB, cbuf);
    }

    // sequential recurrence: 512 launches, h double-buffered
    const __bf16* h_in = hA;
    __bf16*       h_out = hB;
    for (int t = 0; t < T_; ++t) {
        lstm_step_kernel<<<64, 256, 0, stream>>>(xbf + (size_t)t * I_,
                                                 h_in, Wz, bias, cbuf, h_out);
        const __bf16* tmp = h_out;
        h_out = (__bf16*)h_in;
        h_in = tmp;
    }

    // final projection on last hidden state
    fc_kernel<<<64, 256, 0, stream>>>(h_in, Wfcb, bfc, out);
}
